// CFA_4337916969251
// MI455X (gfx1250) — compile-verified, matrix operand order checked by reading, not run on hardware
//
#include <hip/hip_runtime.h>
#include <hip/hip_bf16.h>
#include <stdint.h>

// ---------------------------------------------------------------------------
// MI455X (gfx1250) implementation of the transposed-attention block.
// All heavy GEMMs run through v_wmma_f32_16x16x32_bf16 (wave32).
// Every WMMA operand is staged bf16 row-major-over-K so each lane's fragment
// half is one contiguous 16-byte global_load_b128 (ISA 7.12.2 A layout:
// lanes 0-15: M=lane, K=[0..7],[16..23]; lanes 16-31: K=[8..15],[24..31]).
// Bt (the "B" operand) is loaded with the mirrored layout so D = A * Bt^T.
// The K loop is software-pipelined (two in-flight fragment groups) so the
// backend can overlap the 10 b128 loads of group g1 with the 4 WMMAs of g0
// instead of emitting s_wait_loadcnt 0 before every multiply.
// Requires ws_size >= ~455 MB of scratch.
// ---------------------------------------------------------------------------

typedef __attribute__((ext_vector_type(16))) __bf16 v16bf;
typedef __attribute__((ext_vector_type(8)))  float  v8f;

union FragBF {
  uint4 u[2];   // 2 x 16B contiguous loads
  v16bf v;      // 16 bf16 = 8 VGPRs
};

struct KGroup {
  FragBF a;     // 16(M) x 32(K) A fragment
  FragBF b[4];  // 4 x [16(N) x 32(K)] Bt fragments
};

__device__ __forceinline__ void load_group(KGroup& g,
                                           const __bf16* __restrict__ ap,
                                           const __bf16* __restrict__ bp0,
                                           long ldb, int kb)
{
  g.a.u[0] = *(const uint4*)(ap + kb);
  g.a.u[1] = *(const uint4*)(ap + kb + 16);
#pragma unroll
  for (int t = 0; t < 4; ++t) {
    const __bf16* bp = bp0 + (long)(t * 16) * ldb + kb;
    g.b[t].u[0] = *(const uint4*)(bp);
    g.b[t].u[1] = *(const uint4*)(bp + 16);
  }
}

__device__ __forceinline__ void mma_group(const KGroup& g, v8f acc[4])
{
#pragma unroll
  for (int t = 0; t < 4; ++t) {
    acc[t] = __builtin_amdgcn_wmma_f32_16x16x32_bf16(
        false, g.a.v, false, g.b[t].v, (short)0, acc[t], false, false);
  }
}

// ---------------------------------------------------------------------------
// Generic bf16 WMMA GEMM:  D[m][n] = sum_k A[m][k] * Bt[n][k]  (+ bias[m])
// A: bf16 [M x K] row-major (lda), Bt: bf16 [N x K] row-major (ldb).
// Output either f32 (Df) or bf16 (Dh), row-major [M x N] (ldd).
// Per-z offsets: off = (z/zdiv)*outer + (z%zdiv)*inner (covers b and b*h grids).
// Each wave: 16(M) x 64(N) strip, K-steps of 32 (pipelined by 2).
// Block 256 = 8 waves -> 128(M) x 64(N) per block. K must be a multiple of 64.
// ---------------------------------------------------------------------------
__global__ __launch_bounds__(256) void gemm_wmma_bf16(
    const __bf16* __restrict__ A, long lda, long sAo, long sAi,
    const __bf16* __restrict__ B, long ldb, long sBo, long sBi,
    float* __restrict__ Df, __bf16* __restrict__ Dh,
    long ldd, long sDo, long sDi,
    const float* __restrict__ bias, int K, int zdiv)
{
  const int lane = threadIdx.x & 31;
  const int wave = threadIdx.x >> 5;
  const int zo = blockIdx.z / zdiv;
  const int zi = blockIdx.z - zo * zdiv;
  A += (long)zo * sAo + (long)zi * sAi;
  B += (long)zo * sBo + (long)zi * sBi;
  const long dbase = (long)zo * sDo + (long)zi * sDi;

  const int m0   = (blockIdx.y * 8 + wave) * 16;
  const int n0   = blockIdx.x * 64;
  const int row  = lane & 15;          // M row (A) / N row (Bt) within tile
  const int koff = (lane >> 4) * 8;    // K sub-offset per half-wave

  v8f acc[4] = {};

  const __bf16* ap  = A + (long)(m0 + row) * lda + koff;
  const __bf16* bp0 = B + (long)(n0 + row) * ldb + koff;

  // Software pipeline: group g0 computes while group g1 loads (and vice versa).
  KGroup g0, g1;
  load_group(g0, ap, bp0, ldb, 0);
#pragma unroll 1
  for (int kb = 0; kb < K; kb += 64) {
    load_group(g1, ap, bp0, ldb, kb + 32);
    // Pull the panel a few K-steps ahead into cache (gfx1250 global_prefetch).
    __builtin_prefetch(ap + kb + 256, 0, 1);
    __builtin_prefetch(bp0 + kb + 256, 0, 1);
    mma_group(g0, acc);
    if (kb + 64 < K) load_group(g0, ap, bp0, ldb, kb + 64);
    mma_group(g1, acc);
  }

  // C/D layout: VGPR r -> M = r (lanes 0-15) / r+8 (lanes 16-31); N = lane&15
  const int hi   = (lane >> 4) << 3;
  const int ncol = lane & 15;
#pragma unroll
  for (int r = 0; r < 8; ++r) {
    const int m = m0 + hi + r;
    const float bv = bias ? bias[m] : 0.0f;
#pragma unroll
    for (int t = 0; t < 4; ++t) {
      const float val = acc[t][r] + bv;
      const long off = dbase + (long)m * ldd + (n0 + t * 16 + ncol);
      if (Df) Df[off] = val;
      else    Dh[off] = (__bf16)val;
    }
  }
}

// ---------------------------------------------------------------------------
// Transpose+convert: out_bf16[c_inner][r] = (bf16)(in[r][c_inner] (+ in2))
// 32x32 LDS tiles, coalesced both sides. rows, cols multiples of 32.
// Also used to fuse out + pos_out into the transposed proj input.
// ---------------------------------------------------------------------------
__global__ void transpose_cvt_bf16(const float* __restrict__ in,
                                   const float* __restrict__ in2,
                                   __bf16* __restrict__ out,
                                   long inZ, long outZ, int rows, int cols)
{
  __shared__ float tile[32][33];
  const float* p  = in + (long)blockIdx.z * inZ;
  const float* p2 = in2 ? in2 + (long)blockIdx.z * inZ : nullptr;
  __bf16* op = out + (long)blockIdx.z * outZ;
  const int c0 = blockIdx.x * 32;
  const int r0 = blockIdx.y * 32;
#pragma unroll
  for (int i = 0; i < 4; ++i) {
    const int r = r0 + threadIdx.y + 8 * i;
    float v = p[(long)r * cols + c0 + threadIdx.x];
    if (p2) v += p2[(long)r * cols + c0 + threadIdx.x];
    tile[threadIdx.y + 8 * i][threadIdx.x] = v;
  }
  __syncthreads();
#pragma unroll
  for (int i = 0; i < 4; ++i) {
    const int c = c0 + threadIdx.y + 8 * i;
    op[(long)c * rows + r0 + threadIdx.x] =
        (__bf16)tile[threadIdx.x][threadIdx.y + 8 * i];
  }
}

// Depthwise 3x3, SAME padding, 128x128 images. w: [C][3][3].
__global__ void dwconv3x3_k(const float* __restrict__ in,
                            const float* __restrict__ w,
                            float* __restrict__ out, int C)
{
  const int n  = blockIdx.x * blockDim.x + threadIdx.x;   // 0..16383
  const int c  = blockIdx.y;
  const long base = ((long)blockIdx.z * C + c) * 16384;
  const int hh = n >> 7, ww = n & 127;
  const float* wp = w + c * 9;
  float s = 0.f;
#pragma unroll
  for (int dy = -1; dy <= 1; ++dy) {
    const int y = hh + dy;
    if ((unsigned)y >= 128u) continue;
#pragma unroll
    for (int dx = -1; dx <= 1; ++dx) {
      const int x = ww + dx;
      if ((unsigned)x >= 128u) continue;
      s += wp[(dy + 1) * 3 + (dx + 1)] * in[base + y * 128 + x];
    }
  }
  out[base + n] = s;
}

// Row l2-normalize (over 16384 spatial elems) + convert to bf16.
__global__ __launch_bounds__(256) void l2norm_row(const float* __restrict__ in,
                                                  __bf16* __restrict__ out,
                                                  long inB, long outB)
{
  __shared__ float red[256];
  const float* ip = in  + (long)blockIdx.y * inB  + (long)blockIdx.x * 16384;
  __bf16*      op = out + (long)blockIdx.y * outB + (long)blockIdx.x * 16384;
  float ss = 0.f;
  for (int i = threadIdx.x; i < 16384; i += 256) { const float v = ip[i]; ss += v * v; }
  red[threadIdx.x] = ss; __syncthreads();
  for (int s = 128; s > 0; s >>= 1) {
    if (threadIdx.x < s) red[threadIdx.x] += red[threadIdx.x + s];
    __syncthreads();
  }
  const float scale = 1.0f / fmaxf(sqrtf(red[0]), 1e-12f);
  for (int i = threadIdx.x; i < 16384; i += 256) op[i] = (__bf16)(ip[i] * scale);
}

// Softmax over rows of 128 (channel attention), optional per-head temperature,
// bf16 output (feeds the next WMMA GEMM directly as the A operand).
__global__ __launch_bounds__(128) void softmax128(const float* __restrict__ in,
                                                  __bf16* __restrict__ out,
                                                  const float* __restrict__ temp)
{
  __shared__ float red[128];
  const int row = blockIdx.x;                 // row = (b*4+h)*128 + c
  const float scale = temp ? temp[(row >> 7) & 3] : 1.0f;
  const float v = in[(long)row * 128 + threadIdx.x] * scale;
  red[threadIdx.x] = v; __syncthreads();
  for (int s = 64; s > 0; s >>= 1) {
    if (threadIdx.x < s) red[threadIdx.x] = fmaxf(red[threadIdx.x], red[threadIdx.x + s]);
    __syncthreads();
  }
  const float mx = red[0]; __syncthreads();
  const float e = expf(v - mx);
  red[threadIdx.x] = e; __syncthreads();
  for (int s = 64; s > 0; s >>= 1) {
    if (threadIdx.x < s) red[threadIdx.x] += red[threadIdx.x + s];
    __syncthreads();
  }
  out[(long)row * 128 + threadIdx.x] = (__bf16)(e / red[0]);
}

__global__ void f32_to_bf16(const float* __restrict__ in, __bf16* __restrict__ out, int n)
{
  const int i = blockIdx.x * 256 + threadIdx.x;
  if (i < n) out[i] = (__bf16)in[i];
}

// bias[o] = sum_c W[o][c] * pos_embed[c]   (fold pos_embed into GEMM epilogue)
__global__ void pos_bias(const float* __restrict__ w, const float* __restrict__ pe,
                         float* __restrict__ bias)
{
  const int o = blockIdx.x * 256 + threadIdx.x;
  if (o < 512) {
    float s = 0.f;
    for (int c = 0; c < 512; ++c) s += w[o * 512 + c] * pe[c];
    bias[o] = s;
  }
}

// ---------------------------------------------------------------------------
static void launch_gemm(hipStream_t s,
                        const __bf16* A, long lda, long sAo, long sAi,
                        const __bf16* B, long ldb, long sBo, long sBi,
                        float* Df, __bf16* Dh, long ldd, long sDo, long sDi,
                        const float* bias, int M, int N, int K, int Z, int zdiv)
{
  dim3 grid(N / 64, M / 128, Z);
  gemm_wmma_bf16<<<grid, 256, 0, s>>>(A, lda, sAo, sAi, B, ldb, sBo, sBi,
                                      Df, Dh, ldd, sDo, sDi, bias, K, zdiv);
}

extern "C" void kernel_launch(void* const* d_in, const int* in_sizes, int n_in,
                              void* d_out, int out_size, void* d_ws, size_t ws_size,
                              hipStream_t stream)
{
  (void)in_sizes; (void)n_in; (void)out_size;
  const long N = 16384;            // H*W
  const long C = 512;

  const float* x      = (const float*)d_in[0];
  const float* y      = (const float*)d_in[1];
  const float* temp   = (const float*)d_in[2];   // [4]
  const float* q_w    = (const float*)d_in[3];
  const float* q_dw   = (const float*)d_in[4];
  const float* kv_w   = (const float*)d_in[5];
  const float* kv_dw  = (const float*)d_in[6];
  const float* proj_w = (const float*)d_in[7];
  const float* pe     = (const float*)d_in[8];
  const float* pq_w   = (const float*)d_in[9];
  const float* pk_w   = (const float*)d_in[10];
  float* outp = (float*)d_out;

  // ---- workspace carve-up (deterministic) --------------------------------
  char* ws = (char*)d_ws;
  size_t off = 0;
  auto alloc = [&](size_t bytes) -> char* {
    char* p = ws + off;
    off += (bytes + 255) & ~(size_t)255;
    return p;
  };
  __bf16* xt     = (__bf16*)alloc(2 * N * C * 2);        // x^T  [b][n][c]
  __bf16* yt     = (__bf16*)alloc(2 * N * C * 2);        // y^T
  __bf16* wq     = (__bf16*)alloc(512 * 512 * 2);
  __bf16* wkv    = (__bf16*)alloc(1024 * 512 * 2);
  __bf16* wproj  = (__bf16*)alloc(512 * 512 * 2);
  __bf16* wpq    = (__bf16*)alloc(512 * 512 * 2);
  __bf16* wpk    = (__bf16*)alloc(512 * 512 * 2);
  float*  biasq  = (float*)alloc(512 * 4);
  float*  biask  = (float*)alloc(512 * 4);
  float*  logits = (float*)alloc(8 * 128 * 128 * 4);     // reused attn/pos
  __bf16* prob   = (__bf16*)alloc(8 * 128 * 128 * 2);    // reused attn/pos
  float*  q1f    = (float*)alloc(2 * 512 * N * 4);       // conv1x1(q) f32
  float*  kv1f   = (float*)alloc(2 * 1024 * N * 4);      // conv1x1(kv) f32
  float*  qdf    = (float*)alloc(2 * 512 * N * 4);       // dwconv(q)
  float*  kvdf   = (float*)alloc(2 * 1024 * N * 4);      // dwconv(kv)
  if (off > ws_size) return;  // scratch too small; nothing safe to do

  // buffer reuse (lifetimes disjoint):
  __bf16* qh      = (__bf16*)q1f;                        // q1 dead after dwconv
  __bf16* kh      = (__bf16*)kv1f;                       // kv1 dead after dwconv
  __bf16* vt      = (__bf16*)((char*)kv1f + 33554432);   // kv1 + 32MB
  float*  outb    = qdf;                                 // qd dead after l2norm
  __bf16* pq      = yt;                                  // yt dead after kv GEMM
  __bf16* pk      = (__bf16*)kvdf;                       // kvd dead after kh/vt
  float*  pos_out = (float*)((char*)kvdf + 33554432);
  __bf16* fusedT  = xt;                                  // xt dead after pq/pk

  const dim3 tb(32, 8);

  // 1) weights -> bf16, pos bias
  f32_to_bf16<<<(512 * 512 + 255) / 256, 256, 0, stream>>>(q_w,    wq,    512 * 512);
  f32_to_bf16<<<(1024 * 512 + 255) / 256, 256, 0, stream>>>(kv_w,  wkv,  1024 * 512);
  f32_to_bf16<<<(512 * 512 + 255) / 256, 256, 0, stream>>>(proj_w, wproj, 512 * 512);
  f32_to_bf16<<<(512 * 512 + 255) / 256, 256, 0, stream>>>(pq_w,   wpq,   512 * 512);
  f32_to_bf16<<<(512 * 512 + 255) / 256, 256, 0, stream>>>(pk_w,   wpk,   512 * 512);
  pos_bias<<<2, 256, 0, stream>>>(pq_w, pe, biasq);
  pos_bias<<<2, 256, 0, stream>>>(pk_w, pe, biask);

  // 2) activations -> transposed bf16 panels
  transpose_cvt_bf16<<<dim3(512, 16, 2), tb, 0, stream>>>(x, nullptr, xt, C * N, N * C, 512, 16384);
  transpose_cvt_bf16<<<dim3(512, 16, 2), tb, 0, stream>>>(y, nullptr, yt, C * N, N * C, 512, 16384);

  // 3) conv1x1: q1 = q_w @ x ; kv1 = kv_w @ y
  launch_gemm(stream, wq,  512, 0, 0, xt, 512, N * C, 0,
              q1f, nullptr, N, 512 * N, 0, nullptr, 512, 16384, 512, 2, 1);
  launch_gemm(stream, wkv, 512, 0, 0, yt, 512, N * C, 0,
              kv1f, nullptr, N, 1024 * N, 0, nullptr, 1024, 16384, 512, 2, 1);

  // 4) depthwise 3x3
  dwconv3x3_k<<<dim3(64, 512, 2),  256, 0, stream>>>(q1f,  q_dw,  qdf,  512);
  dwconv3x3_k<<<dim3(64, 1024, 2), 256, 0, stream>>>(kv1f, kv_dw, kvdf, 1024);

  // 5) l2norm rows -> bf16 qh, kh ; transpose v -> vt
  l2norm_row<<<dim3(512, 2), 256, 0, stream>>>(qdf,  qh, 512 * N,  512 * N);
  l2norm_row<<<dim3(512, 2), 256, 0, stream>>>(kvdf, kh, 1024 * N, 512 * N);
  transpose_cvt_bf16<<<dim3(512, 16, 2), tb, 0, stream>>>(kvdf + 512 * N, nullptr, vt,
                                                          1024 * N, N * C, 512, 16384);

  // 6) attn = softmax(temperature * qh @ kh^T)  per (b,h): 8 GEMMs K=16384
  launch_gemm(stream, qh, N, 128 * N, 0, kh, N, 128 * N, 0,
              logits, nullptr, 128, 128 * 128, 0, nullptr, 128, 128, 16384, 8, 1);
  softmax128<<<1024, 128, 0, stream>>>(logits, prob, temp);

  // 7) out = attn @ vh : A=prob[z][128x128], Bt=vt[b][n][c] head slice
  launch_gemm(stream, prob, 128, 4 * 128 * 128, 128 * 128,
              vt, 512, N * C, 128,
              outb, nullptr, N, 512 * N, 128 * N, nullptr, 128, 16384, 128, 8, 4);

  // 8) pos branch: pq/pk = pos_w @ (x+pe), bias folded, bf16 output
  launch_gemm(stream, wpq, 512, 0, 0, xt, 512, N * C, 0,
              nullptr, pq, N, 512 * N, 0, biasq, 512, 16384, 512, 2, 1);
  launch_gemm(stream, wpk, 512, 0, 0, xt, 512, N * C, 0,
              nullptr, pk, N, 512 * N, 0, biask, 512, 16384, 512, 2, 1);
  launch_gemm(stream, pq, N, 128 * N, 0, pk, N, 128 * N, 0,
              logits, nullptr, 128, 128 * 128, 0, nullptr, 128, 128, 16384, 8, 1);
  softmax128<<<1024, 128, 0, stream>>>(logits, prob, nullptr);
  launch_gemm(stream, prob, 128, 4 * 128 * 128, 128 * 128,
              vt, 512, N * C, 128,
              pos_out, nullptr, N, 512 * N, 128 * N, nullptr, 128, 16384, 128, 8, 4);

  // 9) fused = out + pos_out -> transposed bf16 ; final proj to d_out (f32)
  transpose_cvt_bf16<<<dim3(512, 16, 2), tb, 0, stream>>>(outb, pos_out, fusedT,
                                                          512 * N, N * C, 512, 16384);
  launch_gemm(stream, wproj, 512, 0, 0, fusedT, 512, N * C, 0,
              outp, nullptr, N, 512 * N, 0, nullptr, 512, 16384, 512, 2, 1);
}
